// FeatureAttention_22213570855477
// MI455X (gfx1250) — matrix operand order repaired and checked by reading, not compiled, not run on hardware
//
#include <hip/hip_runtime.h>
#include <hip/hip_bf16.h>

// ---------------------------------------------------------------------------
// MI455X (gfx1250) implementation: bf16 WMMA (16x16x32) for all three GEMM
// stages + wave-level flash attention. wave32; fragments built with per-lane
// contiguous 16B/32B loads giving k = 16*half + s in both A and B.
// ---------------------------------------------------------------------------

typedef __attribute__((ext_vector_type(16))) __bf16 v16bf;
typedef __attribute__((ext_vector_type(8)))  __bf16 v8bf;
typedef __attribute__((ext_vector_type(8)))  float  v8f;

union FragBF { v16bf v; v8bf h[2]; };

constexpr int  CB    = 256;    // channels (= INNER)
constexpr int  NN    = 2304;   // H*W
constexpr int  NH    = 8;      // heads
constexpr int  HD    = 32;     // head dim
constexpr int  QKVC  = 768;    // 3*INNER
constexpr float SCALE = 0.17677669529663687f;  // 1/sqrt(32)

__device__ __forceinline__ v8f wmma_bf16(v16bf a, v16bf b, v8f c) {
  // D = A(16x32) * B(32x16) + C, f32 accumulate
  return __builtin_amdgcn_wmma_f32_16x16x32_bf16(
      /*neg_a=*/false, a, /*neg_b=*/false, b,
      /*c_mod=*/(short)0, c, /*reuse_a=*/false, /*reuse_b=*/false);
}

__device__ __forceinline__ v8bf cvt8(v8f x) {
  v8bf r;
#pragma unroll
  for (int e = 0; e < 8; ++e) r[e] = (__bf16)x[e];
  return r;
}

// ---------------------------------------------------------------------------
// Kernel 0: dst[x][y] = (bf16) src[y][x]   (src: rows x cols f32)
// LDS-tiled so both global accesses are coalesced.
// ---------------------------------------------------------------------------
__global__ void transpose_cvt_kernel(const float* __restrict__ src,
                                     __bf16* __restrict__ dst,
                                     int rows, int cols) {
  __shared__ float tile[32][33];
  const int tx = threadIdx.x & 31;
  const int ty = threadIdx.x >> 5;          // 0..7
  const int c0 = blockIdx.x * 32;           // src col tile
  const int r0 = blockIdx.y * 32;           // src row tile
#pragma unroll
  for (int i = 0; i < 4; ++i)
    tile[ty + 8 * i][tx] = src[(size_t)(r0 + ty + 8 * i) * cols + (c0 + tx)];
  __syncthreads();
#pragma unroll
  for (int i = 0; i < 4; ++i)
    dst[(size_t)(c0 + ty + 8 * i) * rows + (r0 + tx)] =
        (__bf16)tile[tx][ty + 8 * i];
}

// ---------------------------------------------------------------------------
// Kernel 1: QKV projection.  X[plane] (2304x256, bf16 row-major) x Wt (768x256
// bf16, stored [j][c]) + bias -> Q/K row-major [plane][part][h][n][d] (bf16)
// and V transposed [plane][h][d][n] (bf16).
// Block = 256 threads = 8 waves, wave tile 32x32, workgroup tile 64x128.
// ---------------------------------------------------------------------------
__global__ void qkv_gemm_kernel(const __bf16* __restrict__ xbf,
                                const __bf16* __restrict__ wt_rgb,
                                const __bf16* __restrict__ wt_ycb,
                                const float* __restrict__ bias_rgb,
                                const float* __restrict__ bias_ycb,
                                __bf16* __restrict__ qk,
                                __bf16* __restrict__ vt) {
  const int lane = threadIdx.x & 31;
  const int wave = threadIdx.x >> 5;        // 0..7
  const int lm = lane & 15;
  const int hb = lane >> 4;                 // lane half
  const int plane = blockIdx.z;             // b*2 + s
  const int s = plane & 1;
  const __bf16* wt  = s ? wt_ycb : wt_rgb;
  const float* bias = s ? bias_ycb : bias_rgb;
  const __bf16* x = xbf + (size_t)plane * NN * CB;

  const int nb = blockIdx.x * 64 + (wave & 1) * 32;
  const int jb = blockIdx.y * 128 + (wave >> 1) * 32;

  v8f acc[2][2] = {};
  for (int cb = 0; cb < CB; cb += 32) {
    FragBF a0, a1;
    const __bf16* ar0 = x + (size_t)(nb + lm) * CB + cb + hb * 16;
    const __bf16* ar1 = x + (size_t)(nb + 16 + lm) * CB + cb + hb * 16;
    a0.h[0] = *(const v8bf*)(ar0);
    a0.h[1] = *(const v8bf*)(ar0 + 8);
    a1.h[0] = *(const v8bf*)(ar1);
    a1.h[1] = *(const v8bf*)(ar1 + 8);
    v16bf b0 = *(const v16bf*)(wt + (size_t)(jb + lm) * CB + cb + 16 * hb);
    v16bf b1 = *(const v16bf*)(wt + (size_t)(jb + 16 + lm) * CB + cb + 16 * hb);
    acc[0][0] = wmma_bf16(a0.v, b0, acc[0][0]);
    acc[0][1] = wmma_bf16(a0.v, b1, acc[0][1]);
    acc[1][0] = wmma_bf16(a1.v, b0, acc[1][0]);
    acc[1][1] = wmma_bf16(a1.v, b1, acc[1][1]);
  }

#pragma unroll
  for (int cI = 0; cI < 2; ++cI) {
    const int j = jb + 16 * cI + lm;        // column in [0,768)
    const float bj = bias[j];
    const int part = j >> 8;                // 0=q 1=k 2=v
    const int h = (j >> 5) & 7;
    const int d = j & 31;
#pragma unroll
    for (int rI = 0; rI < 2; ++rI) {
      const int n0 = nb + 16 * rI + 8 * hb;
      if (part < 2) {
        __bf16* dstp = qk + ((((size_t)plane * 2 + part) * NH + h) * NN) * HD + d;
#pragma unroll
        for (int r = 0; r < 8; ++r)
          dstp[(size_t)(n0 + r) * HD] = (__bf16)(acc[rI][cI][r] + bj);
      } else {
        // V transposed: [plane][h][d][n] -> contiguous over n
        __bf16* dstp = vt + (((size_t)plane * NH + h) * HD + d) * NN + n0;
        v8bf pk;
#pragma unroll
        for (int r = 0; r < 8; ++r) pk[r] = (__bf16)(acc[rI][cI][r] + bj);
        *(v8bf*)dstp = pk;
      }
    }
  }
}

// ---------------------------------------------------------------------------
// Kernel 2: flash cross-attention. One wave = 16 query rows; online softmax
// over 72 key tiles of 32. Row stats live per-lane (row = vgpr + 8*half,
// matching the D-matrix layout); P is re-laid-out via a 1KB/wave LDS region.
// Output: attn[dir*2+b][n][inner] f32.
// ---------------------------------------------------------------------------
__global__ void attn_kernel(const __bf16* __restrict__ qk,
                            const __bf16* __restrict__ vt,
                            float* __restrict__ attn) {
  __shared__ __bf16 plds[4][16][32];
  const int lane = threadIdx.x & 31;
  const int wave = threadIdx.x >> 5;        // 0..3
  const int lm = lane & 15;
  const int hb = lane >> 4;
  const int h = blockIdx.y;
  const int z = blockIdx.z;                 // b*2 + dir
  const int b = z >> 1, dir = z & 1;
  const int plane_q  = b * 2 + dir;         // dir0: rgb queries / ycb kv
  const int plane_kv = b * 2 + (dir ^ 1);
  const int qbase = blockIdx.x * 64 + wave * 16;

  const __bf16* Q = qk + ((((size_t)plane_q * 2 + 0) * NH + h) * NN) * HD;
  const __bf16* K = qk + ((((size_t)plane_kv * 2 + 1) * NH + h) * NN) * HD;
  const __bf16* V = vt + (((size_t)plane_kv * NH + h) * HD) * NN;

  FragBF qa;
  {
    const __bf16* qrow = Q + (size_t)(qbase + lm) * HD + hb * 16;
    qa.h[0] = *(const v8bf*)(qrow);
    qa.h[1] = *(const v8bf*)(qrow + 8);
  }

  v8f o0 = {}, o1 = {};
  float mrow[8], lrow[8];
#pragma unroll
  for (int r = 0; r < 8; ++r) { mrow[r] = -1e30f; lrow[r] = 0.0f; }

  for (int t = 0; t < NN / 32; ++t) {
    const int mb = t * 32;
    v16bf kb0 = *(const v16bf*)(K + (size_t)(mb + lm) * HD + 16 * hb);
    v16bf kb1 = *(const v16bf*)(K + (size_t)(mb + 16 + lm) * HD + 16 * hb);
    if (t + 1 < NN / 32)  // gfx1250 global_prefetch_b8 for next K tile
      __builtin_prefetch(K + (size_t)(mb + 32 + lane) * HD, 0, 3);

    v8f zc = {};
    v8f s0 = wmma_bf16(qa.v, kb0, zc);
    v8f s1 = wmma_bf16(qa.v, kb1, zc);

#pragma unroll
    for (int r = 0; r < 8; ++r) {
      float x0 = s0[r] * SCALE, x1 = s1[r] * SCALE;
      float mx = fmaxf(x0, x1);
      mx = fmaxf(mx, __shfl_xor(mx, 1, 32));
      mx = fmaxf(mx, __shfl_xor(mx, 2, 32));
      mx = fmaxf(mx, __shfl_xor(mx, 4, 32));
      mx = fmaxf(mx, __shfl_xor(mx, 8, 32));   // row-wide within 16-lane half
      float mn = fmaxf(mrow[r], mx);
      float f  = __expf(mrow[r] - mn);
      float p0 = __expf(x0 - mn);
      float p1 = __expf(x1 - mn);
      float rs = p0 + p1;
      rs += __shfl_xor(rs, 1, 32);
      rs += __shfl_xor(rs, 2, 32);
      rs += __shfl_xor(rs, 4, 32);
      rs += __shfl_xor(rs, 8, 32);
      lrow[r] = lrow[r] * f + rs;
      mrow[r] = mn;
      o0[r] *= f; o1[r] *= f;
      const int row = r + 8 * hb;
      plds[wave][row][lm]      = (__bf16)p0;
      plds[wave][row][16 + lm] = (__bf16)p1;
    }
    // Reload P as an A fragment (in-wave LDS ordering; no barrier needed).
    FragBF pa;
    pa.h[0] = *(const v8bf*)(&plds[wave][lm][hb * 16]);
    pa.h[1] = *(const v8bf*)(&plds[wave][lm][8 + hb * 16]);
    v16bf vb0 = *(const v16bf*)(V + (size_t)lm * NN + mb + 16 * hb);
    v16bf vb1 = *(const v16bf*)(V + (size_t)(16 + lm) * NN + mb + 16 * hb);
    o0 = wmma_bf16(pa.v, vb0, o0);
    o1 = wmma_bf16(pa.v, vb1, o1);
  }

  float* outp = attn + (size_t)(dir * 2 + b) * NN * CB;
  const int inner0 = h * HD + lm;
  const int inner1 = h * HD + 16 + lm;
#pragma unroll
  for (int r = 0; r < 8; ++r) {
    const float inv = 1.0f / lrow[r];
    const int n = qbase + 8 * hb + r;
    outp[(size_t)n * CB + inner0] = o0[r] * inv;
    outp[(size_t)n * CB + inner1] = o1[r] * inv;
  }
}

// ---------------------------------------------------------------------------
// Kernel 3: out = (attn_dir0 + attn_dir1) @ W_out + b_out, written transposed
// to [B][C][N]. Same 8-wave / 32x32-per-wave tiling as K1; A is summed+cvt'd
// on the fly; stores are vectorized 32B (accumulator rows are consecutive n).
// ---------------------------------------------------------------------------
__global__ void out_gemm_kernel(const float* __restrict__ attn,
                                const __bf16* __restrict__ wout_t,
                                const float* __restrict__ b_out,
                                float* __restrict__ out) {
  const int lane = threadIdx.x & 31;
  const int wave = threadIdx.x >> 5;
  const int lm = lane & 15, hb = lane >> 4;
  const int b = blockIdx.z;
  const int nb = blockIdx.x * 64 + (wave & 1) * 32;
  const int jb = blockIdx.y * 128 + (wave >> 1) * 32;
  const float* A0 = attn + (size_t)(0 * 2 + b) * NN * CB;
  const float* A1 = attn + (size_t)(1 * 2 + b) * NN * CB;

  v8f acc[2][2] = {};
  for (int cb = 0; cb < CB; cb += 32) {
    FragBF a[2];
#pragma unroll
    for (int rI = 0; rI < 2; ++rI) {
      const size_t base = (size_t)(nb + 16 * rI + lm) * CB + cb + hb * 16;
      v8f u = *(const v8f*)(A0 + base) + *(const v8f*)(A1 + base);
      v8f w = *(const v8f*)(A0 + base + 8) + *(const v8f*)(A1 + base + 8);
      a[rI].h[0] = cvt8(u);
      a[rI].h[1] = cvt8(w);
    }
    v16bf b0 = *(const v16bf*)(wout_t + (size_t)(jb + lm) * CB + cb + 16 * hb);
    v16bf b1 = *(const v16bf*)(wout_t + (size_t)(jb + 16 + lm) * CB + cb + 16 * hb);
    acc[0][0] = wmma_bf16(a[0].v, b0, acc[0][0]);
    acc[0][1] = wmma_bf16(a[0].v, b1, acc[0][1]);
    acc[1][0] = wmma_bf16(a[1].v, b0, acc[1][0]);
    acc[1][1] = wmma_bf16(a[1].v, b1, acc[1][1]);
  }

#pragma unroll
  for (int cI = 0; cI < 2; ++cI) {
    const int c = jb + 16 * cI + lm;
    const float bz = b_out[c];
#pragma unroll
    for (int rI = 0; rI < 2; ++rI) {
      const int n0 = nb + 16 * rI + 8 * hb;
      v8f v = acc[rI][cI];
#pragma unroll
      for (int r = 0; r < 8; ++r) v[r] += bz;
      *(v8f*)(out + ((size_t)b * CB + c) * NN + n0) = v;   // [b][c][n], 32B store
    }
  }
}

// ---------------------------------------------------------------------------
extern "C" void kernel_launch(void* const* d_in, const int* in_sizes, int n_in,
                              void* d_out, int out_size, void* d_ws, size_t ws_size,
                              hipStream_t stream) {
  (void)in_sizes; (void)n_in; (void)out_size; (void)ws_size;
  const float* rgb  = (const float*)d_in[0];
  const float* ycb  = (const float*)d_in[1];
  const float* Wrgb = (const float*)d_in[2];
  const float* brgb = (const float*)d_in[3];
  const float* Wycb = (const float*)d_in[4];
  const float* bycb = (const float*)d_in[5];
  const float* Wout = (const float*)d_in[6];
  const float* bout = (const float*)d_in[7];
  float* out = (float*)d_out;

  char* ws = (char*)d_ws;
  size_t off = 0;
  auto take = [&](size_t bytes) -> void* {
    void* p = ws + off;
    off = (off + bytes + 255) & ~(size_t)255;
    return p;
  };
  __bf16* xbf  = (__bf16*)take((size_t)4 * NN * CB * 2);        // X[b*2+s][n][c]
  __bf16* wtr  = (__bf16*)take((size_t)QKVC * CB * 2);          // W_rgb^T [j][c]
  __bf16* wty  = (__bf16*)take((size_t)QKVC * CB * 2);          // W_ycb^T [j][c]
  __bf16* wot  = (__bf16*)take((size_t)CB * CB * 2);            // W_out^T [c][inner]
  __bf16* qkb  = (__bf16*)take((size_t)4 * 2 * NH * NN * HD * 2);  // [plane][q/k][h][n][d]
  __bf16* vtb  = (__bf16*)take((size_t)4 * NH * HD * NN * 2);   // [plane][h][d][n]
  float*  attb = (float*)take((size_t)4 * NN * CB * 4);         // [dir*2+b][n][inner]

  // Stage 0: transpose + f32->bf16 conversion (activations + weights).
  for (int b = 0; b < 2; ++b) {
    transpose_cvt_kernel<<<dim3(NN / 32, CB / 32), 256, 0, stream>>>(
        rgb + (size_t)b * CB * NN, xbf + (size_t)(b * 2 + 0) * NN * CB, CB, NN);
    transpose_cvt_kernel<<<dim3(NN / 32, CB / 32), 256, 0, stream>>>(
        ycb + (size_t)b * CB * NN, xbf + (size_t)(b * 2 + 1) * NN * CB, CB, NN);
  }
  transpose_cvt_kernel<<<dim3(QKVC / 32, CB / 32), 256, 0, stream>>>(Wrgb, wtr, CB, QKVC);
  transpose_cvt_kernel<<<dim3(QKVC / 32, CB / 32), 256, 0, stream>>>(Wycb, wty, CB, QKVC);
  transpose_cvt_kernel<<<dim3(CB / 32, CB / 32), 256, 0, stream>>>(Wout, wot, CB, CB);

  // Stage 1: fused QKV projections (both streams, both batches via grid.z).
  qkv_gemm_kernel<<<dim3(NN / 64, QKVC / 128, 4), 256, 0, stream>>>(
      xbf, wtr, wty, brgb, bycb, qkb, vtb);

  // Stage 2: flash cross-attention (both directions via grid.z).
  attn_kernel<<<dim3(NN / 64, NH, 4), 128, 0, stream>>>(qkb, vtb, attb);

  // Stage 3: output projection + bias + transpose to [B][C][H][W].
  out_gemm_kernel<<<dim3(NN / 64, CB / 128, 2), 256, 0, stream>>>(
      attb, wot, bout, out);
}
